// EncoderRNN_52132313039323
// MI455X (gfx1250) — matrix operand order repaired and checked by reading, not compiled, not run on hardware
//
#include <hip/hip_runtime.h>

// ---------------- CDNA5 WMMA / TDM types ----------------
typedef __attribute__((ext_vector_type(16))) __bf16 v16bf;
typedef __attribute__((ext_vector_type(8)))  float  v8f;
typedef __attribute__((ext_vector_type(4)))  unsigned su4;
typedef __attribute__((ext_vector_type(8)))  unsigned su8;

#define T_   48
#define B_   128
#define H_   512
#define V_   8000
#define ED_  32
#define JNT_ 576          // H + 2*ED
#define G3H_ 1536         // 3*H
#define TB_  (T_ * B_)    // 6144

__device__ __forceinline__ unsigned short f2bf(float f) {
  unsigned int u = __builtin_bit_cast(unsigned int, f);
  unsigned int r = u + 0x7fffu + ((u >> 16) & 1u);   // round-to-nearest-even
  return (unsigned short)(r >> 16);
}

union FragAB { v16bf v; unsigned int u[8]; };

// ---- fast gate math (CDNA5 native rcp / tanh) ----
__device__ __forceinline__ float fast_rcp(float x) {
#if __has_builtin(__builtin_amdgcn_rcpf)
  return __builtin_amdgcn_rcpf(x);
#else
  return 1.0f / x;
#endif
}
__device__ __forceinline__ float sigmoid_f(float x) {
  return fast_rcp(1.0f + __expf(-x));
}
__device__ __forceinline__ float tanh_f(float x) {
#if __has_builtin(__builtin_amdgcn_tanhf)
  return __builtin_amdgcn_tanhf(x);
#else
  return 1.0f - 2.0f * fast_rcp(__expf(2.0f * x) + 1.0f);
#endif
}

// ---- CDNA5 async / tensor data movers ----
__device__ __forceinline__ unsigned lds_lo32(const void* p) {
  return (unsigned)(unsigned long long)(uintptr_t)p;  // addr[31:0] = LDS offset
}
__device__ __forceinline__ void async_ld_b128(unsigned ldsaddr, const void* g) {
  asm volatile("global_load_async_to_lds_b128 %0, %1, off"
               :: "v"(ldsaddr), "v"((unsigned long long)(uintptr_t)g)
               : "memory");
}
__device__ __forceinline__ void wait_asynccnt0() {
#if __has_builtin(__builtin_amdgcn_s_wait_asynccnt)
  __builtin_amdgcn_s_wait_asynccnt(0);
#else
  asm volatile("s_wait_asynccnt 0x0" ::: "memory");
#endif
}
__device__ __forceinline__ void wait_tensorcnt0() {
#if __has_builtin(__builtin_amdgcn_s_wait_tensorcnt)
  __builtin_amdgcn_s_wait_tensorcnt(0);
#else
  asm volatile("s_wait_tensorcnt 0x0" ::: "memory");
#endif
}

// TDM: 2-D tile load (tile_c elems per row, tile_r rows, row stride in elems,
// 2-byte elements) from global to LDS. One issue per wave; D# per ISA 8.3/8.4.
__device__ __forceinline__ void tdm_load_2d(unsigned lds, const void* g,
                                            unsigned tile_c, unsigned tile_r,
                                            unsigned tensor_d0, unsigned tensor_d1,
                                            unsigned stride) {
  unsigned long long ga = (unsigned long long)(uintptr_t)g;
  su4 g0;
  g0[0] = 1u;                                             // count=1, user mode
  g0[1] = lds;                                            // lds_addr (bytes)
  g0[2] = (unsigned)(ga & 0xffffffffu);                   // global_addr[31:0]
  g0[3] = (unsigned)((ga >> 32) & 0x01ffffffu) | (2u << 30); // ga[56:32] | type=2
  su8 g1;
  g1[0] = (1u << 16);                                     // data_size=1 (2 bytes)
  g1[1] = (tensor_d0 & 0xffffu) << 16;                    // tensor_dim0[15:0]
  g1[2] = (tensor_d0 >> 16) | ((tensor_d1 & 0xffffu) << 16);
  g1[3] = (tensor_d1 >> 16) | ((tile_c & 0xffffu) << 16); // tile_dim0
  g1[4] = (tile_r & 0xffffu);                             // tile_dim1 (tile_dim2=0)
  g1[5] = stride;                                         // dim0_stride[31:0]
  g1[6] = 0u;                                             // stride[47:32], dim1_stride lo
  g1[7] = 0u;
  asm volatile("tensor_load_to_lds %0, %1" :: "s"(g0), "s"(g1) : "memory");
}

// ---------------- elementwise helpers ----------------
__global__ void k_f32_to_bf16(const float* __restrict__ src,
                              unsigned short* __restrict__ dst, int n) {
  int i = blockIdx.x * blockDim.x + threadIdx.x;
  if (i < n) dst[i] = f2bf(src[i]);
}

__global__ void k_zero_h(float* __restrict__ hf, unsigned short* __restrict__ hb, int n) {
  int i = blockIdx.x * blockDim.x + threadIdx.x;
  if (i < n) { hf[i] = 0.0f; hb[i] = 0; }
}

__global__ void k_copy_f32(const float* __restrict__ s, float* __restrict__ d, int n) {
  int i = blockIdx.x * blockDim.x + threadIdx.x;
  if (i < n) d[i] = s[i];
}

// embedding gather + concat -> bf16 joint [T*B, 576]
__global__ void k_embed(const int* __restrict__ seq, const int* __restrict__ c2n,
                        const int* __restrict__ c2f, const float* __restrict__ emb,
                        const float* __restrict__ snd, const float* __restrict__ numw,
                        unsigned short* __restrict__ joint) {
  int tb  = blockIdx.x;
  int idx = seq[tb];
  int s   = c2f[idx];
  int n   = c2n[idx];
  unsigned short* out = joint + (size_t)tb * JNT_;
  for (int c = threadIdx.x; c < H_; c += blockDim.x)
    out[c] = f2bf(emb[(size_t)idx * H_ + c]);
  for (int c = threadIdx.x; c < ED_; c += blockDim.x) {
    out[H_ + c]       = f2bf(snd[s * ED_ + c]);
    out[H_ + ED_ + c] = f2bf(numw[n * ED_ + c]);
  }
}

// ---------------- WMMA GEMM with async/TDM double-buffered staging ----------
// C[M,N] = A[M,K] @ W[N,K]^T + bias[N]   (A, W bf16; C f32 or bf16; opt ReLU)
// Block tile 128x64, 8 waves, each wave 32x32 (2x2 WMMA tiles).
// A tile -> LDS via global_load_async_to_lds_b128 (ASYNCcnt)
// B tile -> LDS via TDM tensor_load_to_lds, wave 0 (TENSORcnt)
template <int OUT_BF16, int RELU>
__global__ __launch_bounds__(256)
void k_gemm(const unsigned short* __restrict__ A,
            const unsigned short* __restrict__ W,
            const float* __restrict__ bias,
            void* __restrict__ Cout, int M, int N, int K) {
  __shared__ __align__(16) unsigned short As[2][128 * 32];
  __shared__ __align__(16) unsigned short Bs[2][64 * 32];
  const int tid  = threadIdx.x;
  const int lane = tid & 31;
  const int wv   = tid >> 5;
  const int wm   = wv >> 1;          // 0..3
  const int wn   = wv & 1;           // 0..1
  const int m0   = blockIdx.y * 128;
  const int n0   = blockIdx.x * 64;
  v8f acc[2][2] = {};

  const int arow = tid >> 1, acol = (tid & 1) * 16;
  auto stage = [&](int buf, int k0) {
    const unsigned short* g = A + (size_t)(m0 + arow) * K + k0 + acol;
    async_ld_b128(lds_lo32(&As[buf][arow * 32 + acol]),     g);
    async_ld_b128(lds_lo32(&As[buf][arow * 32 + acol + 8]), g + 8);
    if (wv == 0)
      tdm_load_2d(lds_lo32(&Bs[buf][0]), W + (size_t)n0 * K + k0,
                  /*tile_c=*/32, /*tile_r=*/64,
                  /*tensor_d0=*/(unsigned)K, /*tensor_d1=*/64,
                  /*stride=*/(unsigned)K);
  };

  const int nk = K / 32;
  stage(0, 0);
  for (int kc = 0; kc < nk; ++kc) {
    const int buf = kc & 1;
    if (wv == 0) wait_tensorcnt0();
    wait_asynccnt0();
    __syncthreads();
    if (kc + 1 < nk) stage(buf ^ 1, (kc + 1) * 32);

    FragAB af[2], bfr[2];
    const int khalfA = ((lane >> 4) & 1) * 8;    // ISA 16-bit A layout
    const int kb0B   = ((lane >> 4) & 1) * 8;    // same k order as A
#pragma unroll
    for (int mi = 0; mi < 2; ++mi) {
      int m = wm * 32 + mi * 16 + (lane & 15);
#pragma unroll
      for (int v = 0; v < 8; ++v) {
        int kb = (v & 3) * 2 + khalfA + (v >> 2) * 16;
        af[mi].u[v] = *(const unsigned int*)&As[buf][m * 32 + kb];
      }
    }
#pragma unroll
    for (int ni = 0; ni < 2; ++ni) {
      int n = wn * 32 + ni * 16 + (lane & 15);
#pragma unroll
      for (int v = 0; v < 8; ++v)
        bfr[ni].u[v] = *(const unsigned int*)&Bs[buf][n * 32 + kb0B + (v & 3) * 2 + (v >> 2) * 16];
    }
#pragma unroll
    for (int mi = 0; mi < 2; ++mi)
#pragma unroll
      for (int ni = 0; ni < 2; ++ni)
        acc[mi][ni] = __builtin_amdgcn_wmma_f32_16x16x32_bf16(
            false, af[mi].v, false, bfr[ni].v, (short)0, acc[mi][ni], false, false);
    __syncthreads();
  }

#pragma unroll
  for (int mi = 0; mi < 2; ++mi) {
#pragma unroll
    for (int ni = 0; ni < 2; ++ni) {
      int n = n0 + wn * 32 + ni * 16 + (lane & 15);
      float bv = bias[n];
#pragma unroll
      for (int i = 0; i < 8; ++i) {
        int m = m0 + wm * 32 + mi * 16 + ((lane >> 4) & 1) * 8 + i;
        float v = acc[mi][ni][i] + bv;
        if (RELU) v = v > 0.0f ? v : 0.0f;
        size_t off = (size_t)m * N + n;
        if (OUT_BF16) ((unsigned short*)Cout)[off] = f2bf(v);
        else          ((float*)Cout)[off] = v;
      }
    }
  }
}

// ---------------- fused GRU step ----------------
// One wave per block computes a 32(batch)x16(hidden) patch: two M-tiles x three
// gate tiles (r,z,n), K=512 -> 96 WMMAs, B fragment reused across both M-tiles.
// grid = (B/32=4, H/16=32), block = 32.
// mode 0: write y bf16 into concat buffer (row stride 1024, base pre-offset)
// mode 1: write y f32 (=)     mode 2: accumulate y f32 (+=)
__global__ __launch_bounds__(32)
void k_gru_step(const float* __restrict__ xw,            // [B,3H] slice at t (incl. bih)
                const unsigned short* __restrict__ hb_in,// [B,H] bf16
                const float* __restrict__ hf_in,         // [B,H] f32
                const unsigned short* __restrict__ Whh,  // [3H,H] bf16 row-major
                const float* __restrict__ bhh,           // [3H]
                float* __restrict__ hf_out,
                unsigned short* __restrict__ hb_out,
                unsigned short* __restrict__ y_bf16,
                float* __restrict__ y_f32,
                int mode) {
  const int lane  = threadIdx.x & 31;
  const int mbase = blockIdx.x * 32;   // two batch tiles
  const int jbase = blockIdx.y * 16;   // hidden tile
  v8f acc[2][3] = {};

  const int khalfA = ((lane >> 4) & 1) * 8;
  const int kb0B   = ((lane >> 4) & 1) * 8;
  for (int k0 = 0; k0 < H_; k0 += 32) {
    FragAB af[2];
#pragma unroll
    for (int mi = 0; mi < 2; ++mi) {
      int m = mbase + mi * 16 + (lane & 15);
#pragma unroll
      for (int v = 0; v < 8; ++v) {
        int kb = (v & 3) * 2 + khalfA + (v >> 2) * 16;
        af[mi].u[v] = *(const unsigned int*)&hb_in[(size_t)m * H_ + k0 + kb];
      }
    }
#pragma unroll
    for (int g = 0; g < 3; ++g) {
      FragAB bfr;
      int n = g * H_ + jbase + (lane & 15);
#pragma unroll
      for (int v = 0; v < 8; ++v)
        bfr.u[v] = *(const unsigned int*)&Whh[(size_t)n * H_ + k0 + kb0B + (v & 3) * 2 + (v >> 2) * 16];
#pragma unroll
      for (int mi = 0; mi < 2; ++mi)
        acc[mi][g] = __builtin_amdgcn_wmma_f32_16x16x32_bf16(
            false, af[mi].v, false, bfr.v, (short)0, acc[mi][g], false, false);
    }
  }

  int jcol = jbase + (lane & 15);
  float br = bhh[jcol], bz = bhh[H_ + jcol], bn = bhh[2 * H_ + jcol];
#pragma unroll
  for (int mi = 0; mi < 2; ++mi) {
#pragma unroll
    for (int i = 0; i < 8; ++i) {
      int m = mbase + mi * 16 + ((lane >> 4) & 1) * 8 + i;
      const float* xwrow = xw + (size_t)m * G3H_;
      float hr = acc[mi][0][i] + br, hz = acc[mi][1][i] + bz, hn = acc[mi][2][i] + bn;
      float xr = xwrow[jcol], xz = xwrow[H_ + jcol], xn = xwrow[2 * H_ + jcol];
      float r  = sigmoid_f(xr + hr);
      float z  = sigmoid_f(xz + hz);
      float nn = tanh_f(xn + r * hn);
      float hp = hf_in[(size_t)m * H_ + jcol];
      float hv = (1.0f - z) * nn + z * hp;
      hf_out[(size_t)m * H_ + jcol] = hv;
      hb_out[(size_t)m * H_ + jcol] = f2bf(hv);
      if (mode == 0)      y_bf16[(size_t)m * 1024 + jcol] = f2bf(hv);
      else if (mode == 1) y_f32[(size_t)m * H_ + jcol]  = hv;
      else                y_f32[(size_t)m * H_ + jcol] += hv;
    }
  }
}

// ---------------- host driver ----------------
extern "C" void kernel_launch(void* const* d_in, const int* in_sizes, int n_in,
                              void* d_out, int out_size, void* d_ws, size_t ws_size,
                              hipStream_t stream) {
  (void)n_in; (void)out_size; (void)ws_size;
  const int*   seq  = (const int*)d_in[0];
  const int*   c2n  = (const int*)d_in[1];
  const int*   c2f  = (const int*)d_in[2];
  const float* emb  = (const float*)d_in[3];
  const float* snd  = (const float*)d_in[4];
  const float* numw = (const float*)d_in[5];
  const float *Wih[2][2], *Whh[2][2], *bih[2][2], *bhh[2][2];
  int WihN[2][2], WhhN[2][2];
  for (int l = 0; l < 2; ++l)
    for (int d = 0; d < 2; ++d) {
      int base = 6 + l * 8 + d * 4;
      Wih[l][d] = (const float*)d_in[base + 0];  WihN[l][d] = in_sizes[base + 0];
      Whh[l][d] = (const float*)d_in[base + 1];  WhhN[l][d] = in_sizes[base + 1];
      bih[l][d] = (const float*)d_in[base + 2];
      bhh[l][d] = (const float*)d_in[base + 3];
    }
  const float* cW1 = (const float*)d_in[22];
  const float* cb1 = (const float*)d_in[23];
  const float* cW2 = (const float*)d_in[24];
  const float* cb2 = (const float*)d_in[25];

  // workspace carve-up (256B aligned)
  char* wsb = (char*)d_ws; size_t off = 0;
  auto alloc = [&](size_t bytes) -> void* {
    void* p = wsb + off; off += (bytes + 255) & ~(size_t)255; return p;
  };
  unsigned short* joint = (unsigned short*)alloc((size_t)TB_ * JNT_ * 2);
  unsigned short* x1    = (unsigned short*)alloc((size_t)TB_ * 1024 * 2);
  unsigned short* outBf = (unsigned short*)alloc((size_t)TB_ * H_ * 2);
  unsigned short* hid1  = (unsigned short*)alloc((size_t)TB_ * H_ * 2);
  float*          xw    = (float*)alloc((size_t)TB_ * G3H_ * 4);
  float*          hF[2] = {(float*)alloc(B_ * H_ * 4), (float*)alloc(B_ * H_ * 4)};
  unsigned short* hB[2] = {(unsigned short*)alloc(B_ * H_ * 2),
                           (unsigned short*)alloc(B_ * H_ * 2)};
  unsigned short *WihB[2][2], *WhhB[2][2];
  for (int l = 0; l < 2; ++l)
    for (int d = 0; d < 2; ++d) {
      WihB[l][d] = (unsigned short*)alloc((size_t)WihN[l][d] * 2);
      WhhB[l][d] = (unsigned short*)alloc((size_t)WhhN[l][d] * 2);
    }
  unsigned short* cW1b = (unsigned short*)alloc((size_t)H_ * H_ * 2);
  unsigned short* cW2b = (unsigned short*)alloc((size_t)V_ * H_ * 2);

  float* outp    = (float*)d_out;                     // [T,B,H]
  float* hiddenp = outp + (size_t)TB_ * H_;           // [4,B,H]
  float* clsp    = hiddenp + (size_t)4 * B_ * H_;     // [T,B,V]

  auto conv = [&](const float* s, unsigned short* d, int n) {
    k_f32_to_bf16<<<(n + 255) / 256, 256, 0, stream>>>(s, d, n);
  };
  for (int l = 0; l < 2; ++l)
    for (int d = 0; d < 2; ++d) {
      conv(Wih[l][d], WihB[l][d], WihN[l][d]);
      conv(Whh[l][d], WhhB[l][d], WhhN[l][d]);
    }
  conv(cW1, cW1b, H_ * H_);
  conv(cW2, cW2b, V_ * H_);

  k_embed<<<TB_, 128, 0, stream>>>(seq, c2n, c2f, emb, snd, numw, joint);

  auto run_dir = [&](int layer, int dir) {
    const unsigned short* Ain = (layer == 0) ? joint : x1;
    int K = (layer == 0) ? JNT_ : 1024;
    k_gemm<0, 0><<<dim3(G3H_ / 64, TB_ / 128), 256, 0, stream>>>(
        Ain, WihB[layer][dir], bih[layer][dir], xw, TB_, G3H_, K);
    k_zero_h<<<(B_ * H_ + 255) / 256, 256, 0, stream>>>(hF[0], hB[0], B_ * H_);
    int cur = 0;
    for (int s = 0; s < T_; ++s) {
      int t = (dir == 0) ? s : (T_ - 1 - s);
      int nxt = cur ^ 1;
      unsigned short* yb = nullptr; float* yf = nullptr; int mode;
      if (layer == 0) { mode = 0; yb = x1 + (size_t)t * B_ * 1024 + dir * H_; }
      else            { mode = (dir == 0) ? 1 : 2; yf = outp + (size_t)t * B_ * H_; }
      k_gru_step<<<dim3(B_ / 32, H_ / 16), 32, 0, stream>>>(
          xw + (size_t)t * B_ * G3H_, hB[cur], hF[cur],
          WhhB[layer][dir], bhh[layer][dir], hF[nxt], hB[nxt], yb, yf, mode);
      cur = nxt;
    }
    k_copy_f32<<<(B_ * H_ + 255) / 256, 256, 0, stream>>>(
        hF[cur], hiddenp + (size_t)(layer * 2 + dir) * B_ * H_, B_ * H_);
  };
  run_dir(0, 0); run_dir(0, 1);
  run_dir(1, 0); run_dir(1, 1);

  // classifier: relu(outputs @ cW1^T + cb1) @ cW2^T + cb2
  k_f32_to_bf16<<<(TB_ * H_ + 255) / 256, 256, 0, stream>>>(outp, outBf, TB_ * H_);
  k_gemm<1, 1><<<dim3(H_ / 64, TB_ / 128), 256, 0, stream>>>(
      outBf, cW1b, cb1, hid1, TB_, H_, H_);
  k_gemm<0, 0><<<dim3(V_ / 64, TB_ / 128), 256, 0, stream>>>(
      hid1, cW2b, cb2, clsp, TB_, V_, H_);
}
